// PerfGAT_89644557402223
// MI455X (gfx1250) — compile-verified, matrix operand order checked by reading, not run on hardware
//
#include <hip/hip_runtime.h>
#include <hip/hip_bf16.h>

typedef __attribute__((ext_vector_type(16))) _Float16 v16h;
typedef __attribute__((ext_vector_type(8)))  float    v8f;

#define N_NODES 4096
#define E_EDGES 131072
#define H_HEADS 3
#define D_OUT   8
#define IN_DIM  256
#define HD      24          // H*D
#define K_DEL_C 26214       // int(0.2*E)
#define K_KEEP_C (E_EDGES - K_DEL_C)        // 104858
#define K_ADD_C 832102      // int(0.05*(N*N-N-E))

// d_out layout (reference return order), float32
#define OFF_OUT_H    0
#define OFF_OUT_KEPT ((size_t)N_NODES*HD)                 // 98304
#define OFF_OUT_ADD  (OFF_OUT_KEPT + (size_t)K_KEEP_C)    // 203162
#define OFF_OUT_U    (OFF_OUT_ADD + (size_t)K_ADD_C)
#define OFF_OUT_V    (OFF_OUT_U  + (size_t)K_ADD_C)

// ---- workspace layout (bytes) ----
#define WS_OFF_X16   ((size_t)0)
#define WS_OFF_W16   (WS_OFF_X16  + (size_t)N_NODES*IN_DIM*2)
#define WS_OFF_FEAT  (WS_OFF_W16  + (size_t)32*IN_DIM*2)          // feat padded [N][32] f32
#define WS_OFF_EL    (WS_OFF_FEAT + (size_t)N_NODES*32*4)
#define WS_OFF_ER    (WS_OFF_EL   + (size_t)N_NODES*H_HEADS*4)
#define WS_OFF_BMP   (WS_OFF_ER   + (size_t)N_NODES*H_HEADS*4)    // adjacency bitmap
#define WS_OFF_MKEY  (WS_OFF_BMP  + (size_t)N_NODES*(N_NODES/8))
#define WS_OFF_SPOS  (WS_OFF_MKEY + (size_t)N_NODES*H_HEADS*4)
#define WS_OFF_RST   (WS_OFF_SPOS + (size_t)N_NODES*H_HEADS*4)
#define WS_OFF_EATT  (WS_OFF_RST  + (size_t)N_NODES*HD*4)
#define WS_OFF_EKEY  (WS_OFF_EATT + (size_t)E_EDGES*4)
#define WS_OFF_HA    (WS_OFF_EKEY + (size_t)E_EDGES*4)            // pos hist hi16
#define WS_OFF_HB    (WS_OFF_HA   + (size_t)65536*4)              // pos hist lo16
#define WS_OFF_EQP   (WS_OFF_HB   + (size_t)65536*4)              // eq-key prefix
#define WS_OFF_KPP   (WS_OFF_EQP  + (size_t)E_EDGES*4)            // keep prefix
#define WS_OFF_BSA   (WS_OFF_KPP  + (size_t)E_EDGES*4)
#define WS_OFF_BSB   (WS_OFF_BSA  + (size_t)512*4)
#define WS_OFF_MNEG  (WS_OFF_BSB  + (size_t)512*4)
#define WS_OFF_SNEG  (WS_OFF_MNEG + (size_t)N_NODES*H_HEADS*4)
#define WS_OFF_NH    (WS_OFF_SNEG + (size_t)N_NODES*H_HEADS*4)    // neg hist hi16
#define WS_OFF_NL    (WS_OFF_NH   + (size_t)65536*4)              // neg hist lo16
#define WS_OFF_PBH   (WS_OFF_NL   + (size_t)65536*4)
#define WS_OFF_PBL   (WS_OFF_PBH  + (size_t)65536*4)
#define WS_OFF_CTH   (WS_OFF_PBL  + (size_t)65536*4)
#define WS_OFF_CTL   (WS_OFF_CTH  + (size_t)65536*4)
#define WS_OFF_SCAL  (WS_OFF_CTL  + (size_t)65536*4)
#define WS_TOTAL     (WS_OFF_SCAL + 256)

__device__ __forceinline__ float lrelu(float x) { return fmaxf(x, 0.2f * x); }
__device__ __forceinline__ unsigned okey(float f) {
    unsigned u = __float_as_uint(f);
    return (u & 0x80000000u) ? ~u : (u | 0x80000000u);
}
__device__ __forceinline__ float ikey(unsigned k) {
    return (k & 0x80000000u) ? __uint_as_float(k & 0x7FFFFFFFu) : __uint_as_float(~k);
}
__device__ __forceinline__ float sigmoidf_(float x) { return 1.f / (1.f + __expf(-x)); }

// ---------------- conversion f32 -> f16 (x and padded fc_w) ----------------
__global__ void k_cvt(const float* __restrict__ x, const float* __restrict__ w,
                      _Float16* __restrict__ x16, _Float16* __restrict__ w16) {
    int i = blockIdx.x * 256 + threadIdx.x;          // grid covers N*IN = 1048576
    if (i < N_NODES * IN_DIM) x16[i] = (_Float16)x[i];
    if (i < 32 * IN_DIM) {
        int n = i >> 8, k = i & 255;
        w16[i] = (n < HD) ? (_Float16)w[n * IN_DIM + k] : (_Float16)0.f;
    }
}

// ---------------- WMMA GEMM: featPad[N][32] = x[N][256] @ fc_w^T ----------------
// one wave per 16x16 C tile; K loop 256 in steps of 32
__global__ __launch_bounds__(32) void k_gemm_wmma(const _Float16* __restrict__ x16,
                                                  const _Float16* __restrict__ w16,
                                                  float* __restrict__ featPad) {
    int tileN = blockIdx.x;          // 0..1  (cols 0..15 / 16..31)
    int tileM = blockIdx.y;          // 0..255
    int lane  = threadIdx.x;         // 0..31
    int row   = tileM * 16 + (lane & 15);
    int col   = tileN * 16 + (lane & 15);
    int kb    = (lane >> 4) * 8;     // lanes 0-15: k 0..7,16..23 ; lanes 16-31: k 8..15,24..31
    v8f c = {};
    for (int k0 = 0; k0 < IN_DIM; k0 += 32) {
        v16h a, b;
        const _Float16* ax = x16 + row * IN_DIM + k0 + kb;
        const _Float16* bw = w16 + col * IN_DIM + k0 + kb;
#pragma unroll
        for (int t = 0; t < 8; ++t) {
            a[t] = ax[t];  a[t + 8] = ax[t + 16];
            b[t] = bw[t];  b[t + 8] = bw[t + 16];
        }
        c = __builtin_amdgcn_wmma_f32_16x16x32_f16(false, a, false, b, (short)0, c, false, false);
    }
#pragma unroll
    for (int r = 0; r < 8; ++r) {
        int m = r + (lane >> 4) * 8;
        featPad[(size_t)(tileM * 16 + m) * 32 + col] = c[r];
    }
}

// ---------------- el/er per (node, head) ----------------
__global__ void k_elr(const float* __restrict__ featPad, const float* __restrict__ attn_l,
                      const float* __restrict__ attn_r, float* __restrict__ el,
                      float* __restrict__ er) {
    int idx = blockIdx.x * 256 + threadIdx.x;        // N*H = 12288
    if (idx >= N_NODES * H_HEADS) return;
    int n = idx / H_HEADS, h = idx % H_HEADS;
    float sl = 0.f, sr = 0.f;
#pragma unroll
    for (int d = 0; d < D_OUT; ++d) {
        float f = featPad[(size_t)n * 32 + h * 8 + d];
        sl += f * attn_l[h * 8 + d];
        sr += f * attn_r[h * 8 + d];
    }
    el[idx] = sl; er[idx] = sr;
}

// ---------------- adjacency bitmap: bit (u*N+v) ----------------
__global__ void k_adj(const int* __restrict__ U, const int* __restrict__ V,
                      unsigned* __restrict__ bmp) {
    int i = blockIdx.x * 256 + threadIdx.x;
    int u = U[i], v = V[i];
    atomicOr(&bmp[(unsigned)u * (N_NODES / 32) + (v >> 5)], 1u << (v & 31));
}

// ---------------- positive graph: segment max / sum / aggregate ----------------
__global__ void k_pos1(const int* __restrict__ U, const int* __restrict__ V,
                       const float* __restrict__ el, const float* __restrict__ er,
                       unsigned* __restrict__ mkey) {
    int i = blockIdx.x * 256 + threadIdx.x;
    int u = U[i], v = V[i];
#pragma unroll
    for (int h = 0; h < H_HEADS; ++h) {
        float e = lrelu(el[u * 3 + h] + er[v * 3 + h]);
        atomicMax(&mkey[v * 3 + h], okey(e));
    }
}
__global__ void k_pos2(const int* __restrict__ U, const int* __restrict__ V,
                       const float* __restrict__ el, const float* __restrict__ er,
                       const unsigned* __restrict__ mkey, float* __restrict__ spos) {
    int i = blockIdx.x * 256 + threadIdx.x;
    int u = U[i], v = V[i];
#pragma unroll
    for (int h = 0; h < H_HEADS; ++h) {
        float e = lrelu(el[u * 3 + h] + er[v * 3 + h]);
        float m = ikey(mkey[v * 3 + h]);
        atomicAdd(&spos[v * 3 + h], __expf(e - m));
    }
}
__global__ void k_pos3(const int* __restrict__ U, const int* __restrict__ V,
                       const float* __restrict__ el, const float* __restrict__ er,
                       const unsigned* __restrict__ mkey, const float* __restrict__ spos,
                       const float* __restrict__ featPad, float* __restrict__ rst,
                       float* __restrict__ eatt, unsigned* __restrict__ ekey,
                       unsigned* __restrict__ hist) {
    int i = blockIdx.x * 256 + threadIdx.x;
    int u = U[i], v = V[i];
    float alpha[H_HEADS], asum = 0.f;
#pragma unroll
    for (int h = 0; h < H_HEADS; ++h) {
        float e = lrelu(el[u * 3 + h] + er[v * 3 + h]);
        float m = ikey(mkey[v * 3 + h]);
        alpha[h] = __expf(e - m) / spos[v * 3 + h];
        asum += alpha[h];
    }
#pragma unroll
    for (int h = 0; h < H_HEADS; ++h)
#pragma unroll
        for (int d = 0; d < D_OUT; ++d)
            atomicAdd(&rst[(size_t)v * HD + h * 8 + d],
                      alpha[h] * featPad[(size_t)u * 32 + h * 8 + d]);
    float att = sigmoidf_(asum * (1.f / 3.f));
    eatt[i] = att;
    unsigned k = okey(att);
    ekey[i] = k;
    atomicAdd(&hist[k >> 16], 1u);
}

// ---------------- h = relu(rst + bias) ----------------
__global__ void k_hout(const float* __restrict__ rst, const float* __restrict__ bias,
                       float* __restrict__ out) {
    int idx = blockIdx.x * 256 + threadIdx.x;        // N*HD = 98304
    if (idx >= N_NODES * HD) return;
    out[OFF_OUT_H + idx] = fmaxf(rst[idx] + bias[idx % HD], 0.f);
}

// ---------------- ascending K-th smallest via 2-level hist ----------------
// writes scal[outIdx] = bin T, scal[outIdx+1] = count strictly below T
__global__ void k_find_asc(const unsigned* __restrict__ hist, unsigned* __restrict__ scal,
                           int outIdx, unsigned targetBase, int subIdx) {
    __shared__ unsigned csum[256];
    int t = threadIdx.x;
    unsigned s = 0;
    for (int b = 0; b < 256; ++b) s += hist[t * 256 + b];
    csum[t] = s;
    __syncthreads();
    if (t == 0) {
        unsigned target = targetBase - ((subIdx >= 0) ? scal[subIdx] : 0u);
        unsigned run = 0; int T = 0; unsigned below = 0;
        for (int c = 0; c < 256; ++c) {
            if (run + csum[c] >= target) {
                for (int b = c * 256;; ++b) {
                    unsigned hv = hist[b];
                    if (run + hv >= target) { T = b; below = run; break; }
                    run += hv;
                }
                break;
            }
            run += csum[c];
        }
        scal[outIdx] = (unsigned)T;
        scal[outIdx + 1] = below;
    }
}

// positive lo-16 histogram restricted to the threshold hi-bin
__global__ void k_pos_hist_lo(const unsigned* __restrict__ ekey, const unsigned* __restrict__ scal,
                              unsigned* __restrict__ histlo) {
    int i = blockIdx.x * 256 + threadIdx.x;
    unsigned k = ekey[i];
    if ((k >> 16) == scal[0]) atomicAdd(&histlo[k & 0xFFFFu], 1u);
}

// ---------------- scan machinery over E (512 blocks x 256) ----------------
__global__ void k_scan_eq(const unsigned* __restrict__ ekey, const unsigned* __restrict__ scal,
                          int* __restrict__ pref, int* __restrict__ bsum) {
    __shared__ int tmp[256];
    int t = threadIdx.x, i = blockIdx.x * 256 + t;
    unsigned thr = (scal[0] << 16) | scal[12];
    int f = (ekey[i] == thr) ? 1 : 0;
    tmp[t] = f; __syncthreads();
    for (int off = 1; off < 256; off <<= 1) {
        int v = (t >= off) ? tmp[t - off] : 0;
        __syncthreads(); tmp[t] += v; __syncthreads();
    }
    pref[i] = tmp[t] - f;
    if (t == 255) bsum[blockIdx.x] = tmp[t];
}
__global__ void k_scan_bsums(int* __restrict__ bsum, int nb) {
    if (threadIdx.x == 0 && blockIdx.x == 0) {
        int run = 0;
        for (int b = 0; b < nb; ++b) { int v = bsum[b]; bsum[b] = run; run += v; }
    }
}
__device__ __forceinline__ int keep_flag(unsigned k, unsigned thr, unsigned need, int eqr) {
    return (k > thr) || (k == thr && (unsigned)eqr >= need);
}
__global__ void k_scan_keep(const unsigned* __restrict__ ekey, const unsigned* __restrict__ scal,
                            const int* __restrict__ eqpref, const int* __restrict__ eqbsum,
                            int* __restrict__ pref, int* __restrict__ bsum) {
    __shared__ int tmp[256];
    int t = threadIdx.x, i = blockIdx.x * 256 + t;
    unsigned thr = (scal[0] << 16) | scal[12];
    unsigned need = (unsigned)K_DEL_C - scal[1] - scal[13];
    int f = keep_flag(ekey[i], thr, need, eqpref[i] + eqbsum[blockIdx.x]);
    tmp[t] = f; __syncthreads();
    for (int off = 1; off < 256; off <<= 1) {
        int v = (t >= off) ? tmp[t - off] : 0;
        __syncthreads(); tmp[t] += v; __syncthreads();
    }
    pref[i] = tmp[t] - f;
    if (t == 255) bsum[blockIdx.x] = tmp[t];
}
__global__ void k_scatter_kept(const unsigned* __restrict__ ekey, const float* __restrict__ eatt,
                               const unsigned* __restrict__ scal, const int* __restrict__ eqpref,
                               const int* __restrict__ eqbsum, const int* __restrict__ kpref,
                               const int* __restrict__ kbsum, float* __restrict__ out) {
    int i = blockIdx.x * 256 + threadIdx.x;
    unsigned thr = (scal[0] << 16) | scal[12];
    unsigned need = (unsigned)K_DEL_C - scal[1] - scal[13];
    if (keep_flag(ekey[i], thr, need, eqpref[i] + eqbsum[blockIdx.x])) {
        int pos = kpref[i] + kbsum[blockIdx.x];
        if (pos < K_KEEP_C) out[OFF_OUT_KEPT + pos] = eatt[i];
    }
}

// ---------------- complement graph: per-column masked max & sum-exp ----------------
__global__ __launch_bounds__(256) void k_neg_ms(const float* __restrict__ el,
                                                const float* __restrict__ er,
                                                const unsigned* __restrict__ bmp,
                                                float* __restrict__ mneg, float* __restrict__ sneg) {
    __shared__ float sel[N_NODES * H_HEADS];
    __shared__ float red[H_HEADS * 256];
    int t = threadIdx.x, j = blockIdx.x;
    for (int i = t; i < N_NODES * H_HEADS; i += 256) sel[i] = el[i];
    __syncthreads();
    float erj[3];
#pragma unroll
    for (int h = 0; h < 3; ++h) erj[h] = er[j * 3 + h];
    int wsh = j >> 5; unsigned wbit = 1u << (j & 31);
    float mx[3] = {-1e30f, -1e30f, -1e30f};
    for (int i = t; i < N_NODES; i += 256) {
        unsigned w = bmp[i * (N_NODES / 32) + wsh];
        if ((w & wbit) || i == j) continue;
#pragma unroll
        for (int h = 0; h < 3; ++h) mx[h] = fmaxf(mx[h], sel[i * 3 + h]);
    }
#pragma unroll
    for (int h = 0; h < 3; ++h) red[h * 256 + t] = mx[h];
    __syncthreads();
    for (int off = 128; off > 0; off >>= 1) {
        if (t < off)
#pragma unroll
            for (int h = 0; h < 3; ++h)
                red[h * 256 + t] = fmaxf(red[h * 256 + t], red[h * 256 + t + off]);
        __syncthreads();
    }
    float mh[3];
#pragma unroll
    for (int h = 0; h < 3; ++h) mh[h] = lrelu(red[h * 256] + erj[h]);  // lrelu monotone
    __syncthreads();
    float sm[3] = {0.f, 0.f, 0.f};
    for (int i = t; i < N_NODES; i += 256) {
        unsigned w = bmp[i * (N_NODES / 32) + wsh];
        if ((w & wbit) || i == j) continue;
#pragma unroll
        for (int h = 0; h < 3; ++h) sm[h] += __expf(lrelu(sel[i * 3 + h] + erj[h]) - mh[h]);
    }
#pragma unroll
    for (int h = 0; h < 3; ++h) red[h * 256 + t] = sm[h];
    __syncthreads();
    for (int off = 128; off > 0; off >>= 1) {
        if (t < off)
#pragma unroll
            for (int h = 0; h < 3; ++h)
                red[h * 256 + t] += red[h * 256 + t + off];
        __syncthreads();
    }
    if (t < 3) { mneg[j * 3 + t] = mh[t]; sneg[j * 3 + t] = red[t * 256]; }
}

// mode 0: hi-16 histogram of val bits; mode 1: lo-16 histogram within scal[4] bin
__global__ __launch_bounds__(256) void k_neg_hist(const float* __restrict__ el,
                                                  const float* __restrict__ er,
                                                  const unsigned* __restrict__ bmp,
                                                  const float* __restrict__ mneg,
                                                  const float* __restrict__ sneg,
                                                  const unsigned* __restrict__ scal,
                                                  unsigned* __restrict__ hist, int mode) {
    __shared__ float sel[N_NODES * H_HEADS];
    int t = threadIdx.x, j = blockIdx.x;
    for (int i = t; i < N_NODES * H_HEADS; i += 256) sel[i] = el[i];
    __syncthreads();
    float erj[3], mh[3], inv[3];
#pragma unroll
    for (int h = 0; h < 3; ++h) {
        erj[h] = er[j * 3 + h];
        mh[h] = mneg[j * 3 + h];
        inv[h] = 1.f / sneg[j * 3 + h];
    }
    int wsh = j >> 5; unsigned wbit = 1u << (j & 31);
    unsigned Thi = (mode == 1) ? scal[4] : 0u;
    for (int i = t; i < N_NODES; i += 256) {
        unsigned w = bmp[i * (N_NODES / 32) + wsh];
        if ((w & wbit) || i == j) continue;
        float a = 0.f;
#pragma unroll
        for (int h = 0; h < 3; ++h)
            a += __expf(lrelu(sel[i * 3 + h] + erj[h]) - mh[h]) * inv[h];
        unsigned k = __float_as_uint(a * (1.f / 3.f));   // positive -> bit order = value order
        if (mode == 0) atomicAdd(&hist[k >> 16], 1u);
        else if ((k >> 16) == Thi) atomicAdd(&hist[k & 0xFFFFu], 1u);
    }
}

// descending K-th largest + per-bin position bases
// scal[outIdx]=T, scal[outIdx+1]=count strictly above, scal[outIdx+2]=remaining in bin T
__global__ void k_find_desc(const unsigned* __restrict__ hist, unsigned* __restrict__ pbase,
                            unsigned* __restrict__ scal, int outIdx, unsigned targetBase,
                            int subIdx) {
    __shared__ unsigned csum[256];
    __shared__ unsigned cabove[256];
    int t = threadIdx.x;
    unsigned s = 0;
    for (int b = 0; b < 256; ++b) s += hist[t * 256 + b];
    csum[t] = s;
    __syncthreads();
    if (t == 0) {
        unsigned run = 0;
        for (int c = 255; c >= 0; --c) { cabove[c] = run; run += csum[c]; }
    }
    __syncthreads();
    unsigned run = cabove[t];
    for (int b = t * 256 + 255; b >= t * 256; --b) { pbase[b] = run; run += hist[b]; }
    if (t == 0) {
        unsigned target = targetBase - ((subIdx >= 0) ? scal[subIdx] : 0u);
        unsigned run2 = 0; int T = 0; unsigned above = 0;
        for (int c = 255; c >= 0; --c) {
            if (run2 + csum[c] >= target) {
                for (int b = c * 256 + 255;; --b) {
                    unsigned hv = hist[b];
                    if (run2 + hv >= target) { T = b; above = run2; break; }
                    run2 += hv;
                }
                break;
            }
            run2 += csum[c];
        }
        scal[outIdx] = (unsigned)T;
        scal[outIdx + 1] = above;
        scal[outIdx + 2] = target - above;
    }
}

// collect top-K complement edges
__global__ __launch_bounds__(256) void k_neg_collect(const float* __restrict__ el,
                                                     const float* __restrict__ er,
                                                     const unsigned* __restrict__ bmp,
                                                     const float* __restrict__ mneg,
                                                     const float* __restrict__ sneg,
                                                     unsigned* __restrict__ scal,
                                                     const unsigned* __restrict__ pbh,
                                                     const unsigned* __restrict__ pbl,
                                                     unsigned* __restrict__ ctrh,
                                                     unsigned* __restrict__ ctrl,
                                                     float* __restrict__ out) {
    __shared__ float sel[N_NODES * H_HEADS];
    int t = threadIdx.x, j = blockIdx.x;
    for (int i = t; i < N_NODES * H_HEADS; i += 256) sel[i] = el[i];
    __syncthreads();
    float erj[3], mh[3], inv[3];
#pragma unroll
    for (int h = 0; h < 3; ++h) {
        erj[h] = er[j * 3 + h];
        mh[h] = mneg[j * 3 + h];
        inv[h] = 1.f / sneg[j * 3 + h];
    }
    int wsh = j >> 5; unsigned wbit = 1u << (j & 31);
    unsigned Thi = scal[4], above_hi = scal[5];
    unsigned Tlo = scal[7], above_lo = scal[8], take_eq = scal[9];
    for (int i = t; i < N_NODES; i += 256) {
        unsigned w = bmp[i * (N_NODES / 32) + wsh];
        if ((w & wbit) || i == j) continue;
        float a = 0.f;
#pragma unroll
        for (int h = 0; h < 3; ++h)
            a += __expf(lrelu(sel[i * 3 + h] + erj[h]) - mh[h]) * inv[h];
        float val = a * (1.f / 3.f);
        unsigned k = __float_as_uint(val);
        unsigned hi = k >> 16, lo = k & 0xFFFFu;
        unsigned pos = 0; bool ok = false;
        if (hi > Thi) {
            pos = pbh[hi] + atomicAdd(&ctrh[hi], 1u); ok = true;
        } else if (hi == Thi) {
            if (lo > Tlo) {
                pos = above_hi + pbl[lo] + atomicAdd(&ctrl[lo], 1u); ok = true;
            } else if (lo == Tlo) {
                unsigned idx = atomicAdd(&scal[11], 1u);
                if (idx < take_eq) { pos = above_hi + above_lo + idx; ok = true; }
            }
        }
        if (ok && pos < (unsigned)K_ADD_C) {
            out[OFF_OUT_ADD + pos] = sigmoidf_(val);
            out[OFF_OUT_U + pos] = (float)i;
            out[OFF_OUT_V + pos] = (float)j;
        }
    }
}

extern "C" void kernel_launch(void* const* d_in, const int* in_sizes, int n_in,
                              void* d_out, int out_size, void* d_ws, size_t ws_size,
                              hipStream_t stream) {
    (void)in_sizes; (void)n_in; (void)out_size; (void)ws_size;
    const float* x      = (const float*)d_in[0];
    const int*   U      = (const int*)d_in[1];
    const int*   V      = (const int*)d_in[2];
    const float* fc_w   = (const float*)d_in[3];
    const float* attn_l = (const float*)d_in[4];
    const float* attn_r = (const float*)d_in[5];
    const float* bias   = (const float*)d_in[6];
    float* out = (float*)d_out;

    char* ws = (char*)d_ws;
    _Float16* x16   = (_Float16*)(ws + WS_OFF_X16);
    _Float16* w16   = (_Float16*)(ws + WS_OFF_W16);
    float* featPad  = (float*)(ws + WS_OFF_FEAT);
    float* el       = (float*)(ws + WS_OFF_EL);
    float* er       = (float*)(ws + WS_OFF_ER);
    unsigned* bmp   = (unsigned*)(ws + WS_OFF_BMP);
    unsigned* mkey  = (unsigned*)(ws + WS_OFF_MKEY);
    float* spos     = (float*)(ws + WS_OFF_SPOS);
    float* rst      = (float*)(ws + WS_OFF_RST);
    float* eatt     = (float*)(ws + WS_OFF_EATT);
    unsigned* ekey  = (unsigned*)(ws + WS_OFF_EKEY);
    unsigned* histA = (unsigned*)(ws + WS_OFF_HA);
    unsigned* histB = (unsigned*)(ws + WS_OFF_HB);
    int* eqpref     = (int*)(ws + WS_OFF_EQP);
    int* kpref      = (int*)(ws + WS_OFF_KPP);
    int* bsumA      = (int*)(ws + WS_OFF_BSA);
    int* bsumB      = (int*)(ws + WS_OFF_BSB);
    float* mneg     = (float*)(ws + WS_OFF_MNEG);
    float* sneg     = (float*)(ws + WS_OFF_SNEG);
    unsigned* histNH= (unsigned*)(ws + WS_OFF_NH);
    unsigned* histNL= (unsigned*)(ws + WS_OFF_NL);
    unsigned* pbh   = (unsigned*)(ws + WS_OFF_PBH);
    unsigned* pbl   = (unsigned*)(ws + WS_OFF_PBL);
    unsigned* ctrh  = (unsigned*)(ws + WS_OFF_CTH);
    unsigned* ctrl  = (unsigned*)(ws + WS_OFF_CTL);
    unsigned* scal  = (unsigned*)(ws + WS_OFF_SCAL);

    hipMemsetAsync(d_ws, 0, WS_TOTAL, stream);       // zero bitmap/hists/sums/counters; mkey=0 is -inf key

    k_cvt<<<4096, 256, 0, stream>>>(x, fc_w, x16, w16);
    k_gemm_wmma<<<dim3(2, 256), 32, 0, stream>>>(x16, w16, featPad);
    k_elr<<<48, 256, 0, stream>>>(featPad, attn_l, attn_r, el, er);
    k_adj<<<512, 256, 0, stream>>>(U, V, bmp);

    k_pos1<<<512, 256, 0, stream>>>(U, V, el, er, mkey);
    k_pos2<<<512, 256, 0, stream>>>(U, V, el, er, mkey, spos);
    k_pos3<<<512, 256, 0, stream>>>(U, V, el, er, mkey, spos, featPad, rst, eatt, ekey, histA);
    k_hout<<<384, 256, 0, stream>>>(rst, bias, out);

    // edge removal threshold (exact 32-bit key + deterministic index tie-break)
    k_find_asc<<<1, 256, 0, stream>>>(histA, scal, 0, (unsigned)K_DEL_C, -1);
    k_pos_hist_lo<<<512, 256, 0, stream>>>(ekey, scal, histB);
    k_find_asc<<<1, 256, 0, stream>>>(histB, scal, 12, (unsigned)K_DEL_C, 1);
    k_scan_eq<<<512, 256, 0, stream>>>(ekey, scal, eqpref, bsumA);
    k_scan_bsums<<<1, 32, 0, stream>>>(bsumA, 512);
    k_scan_keep<<<512, 256, 0, stream>>>(ekey, scal, eqpref, bsumA, kpref, bsumB);
    k_scan_bsums<<<1, 32, 0, stream>>>(bsumB, 512);
    k_scatter_kept<<<512, 256, 0, stream>>>(ekey, eatt, scal, eqpref, bsumA, kpref, bsumB, out);

    // complement graph
    k_neg_ms<<<N_NODES, 256, 0, stream>>>(el, er, bmp, mneg, sneg);
    k_neg_hist<<<N_NODES, 256, 0, stream>>>(el, er, bmp, mneg, sneg, scal, histNH, 0);
    k_find_desc<<<1, 256, 0, stream>>>(histNH, pbh, scal, 4, (unsigned)K_ADD_C, -1);
    k_neg_hist<<<N_NODES, 256, 0, stream>>>(el, er, bmp, mneg, sneg, scal, histNL, 1);
    k_find_desc<<<1, 256, 0, stream>>>(histNL, pbl, scal, 7, (unsigned)K_ADD_C, 5);
    k_neg_collect<<<N_NODES, 256, 0, stream>>>(el, er, bmp, mneg, sneg, scal, pbh, pbl,
                                               ctrh, ctrl, out);
}
